// RelGraphConvLayer_4363686773567
// MI455X (gfx1250) — matrix operand order repaired and checked by reading, not compiled, not run on hardware
//
#include <hip/hip_runtime.h>
#include <hip/hip_bf16.h>

typedef __bf16 bf16_t;
typedef __attribute__((ext_vector_type(16))) __bf16 v16bf;
typedef __attribute__((ext_vector_type(8)))  __bf16 v8bf;
typedef __attribute__((ext_vector_type(8)))  float  v8f;

#define NN 50000
#define EE 600000
#define DD 128
#define OO 128
#define RR 8
#define BB 4

// ---------------------------------------------------------------- fill
__global__ void fill_kernel(float* __restrict__ p, float v, int n) {
    int i = blockIdx.x * blockDim.x + threadIdx.x;
    if (i < n) p[i] = v;
}

// ----------------------------------------------- pre-split X -> bf16 hi/lo
// x [N][D] fp32 -> Xh/Xl [N][D] bf16 (hi / residual-lo).  4 elems / thread.
__global__ void xsplit_kernel(const float* __restrict__ x,
                              bf16_t* __restrict__ Xh, bf16_t* __restrict__ Xl,
                              int n4) {
    int i = blockIdx.x * blockDim.x + threadIdx.x;
    if (i >= n4) return;
    float4 f = *(const float4*)(x + (size_t)i * 4);
    float fv[4] = {f.x, f.y, f.z, f.w};
    bf16_t h[4], l[4];
#pragma unroll
    for (int k = 0; k < 4; ++k) {
        h[k] = (bf16_t)fv[k];
        l[k] = (bf16_t)(fv[k] - (float)h[k]);
    }
    *(ulong1*)(Xh + (size_t)i * 4) = *(ulong1*)h;   // 8-byte stores
    *(ulong1*)(Xl + (size_t)i * 4) = *(ulong1*)l;
}

// ------------------------------------------------- prep: split + transpose
// basis_v [B][D][O] fp32  ->  Bh/Bl [B][O][D] bf16 (hi / residual-lo)
// w_loop  [D][O]    fp32  ->  Lh/Ll [O][D]    bf16
__global__ void prep_kernel(const float* __restrict__ basis_v,
                            const float* __restrict__ w_loop,
                            bf16_t* __restrict__ Bh, bf16_t* __restrict__ Bl,
                            bf16_t* __restrict__ Lh, bf16_t* __restrict__ Ll) {
    int t = blockIdx.x * blockDim.x + threadIdx.x;   // 0 .. D*O-1
    if (t >= DD * OO) return;
    int m = blockIdx.y;                              // 0..B  (B = self-loop)
    int d = t & (DD - 1);
    int o = t >> 7;
    float f;
    bf16_t *ph, *pl;
    if (m < BB) {
        f  = basis_v[((size_t)m * DD + d) * OO + o];
        ph = Bh + ((size_t)m * OO + o) * DD + d;
        pl = Bl + ((size_t)m * OO + o) * DD + d;
    } else {
        f  = w_loop[(size_t)d * OO + o];
        ph = Lh + (size_t)o * DD + d;
        pl = Ll + (size_t)o * DD + d;
    }
    bf16_t h = (bf16_t)f;
    bf16_t l = (bf16_t)(f - (float)h);
    *ph = h;
    *pl = l;
}

// ---------------------------------------------------------------- degree
__global__ void deg_kernel(const int* __restrict__ rel, const int* __restrict__ dst,
                           float* __restrict__ deg, int e) {
    int i = blockIdx.x * blockDim.x + threadIdx.x;
    if (i < e) atomicAdd(&deg[(size_t)rel[i] * NN + dst[i]], 1.0f);
}

// ------------------------------------- basis-fused bf16x3 WMMA GEMM
// y_b = X @ V_b for NB bases (f32 accum, 3 bf16 WMMAs per K-step = hi*hi +
// hi*lo + lo*hi), epilogue emits out[r] = sum_b coeffs[r*NB+b] * y_b.
// Block = 256 threads = 8 waves; wave w owns O-tile w; block owns a 16-row
// M-tile.  D = O = 128 fixed.
template <int NB, int NR>
__global__ void gemm_basis_kernel(const bf16_t* __restrict__ Xh,
                                  const bf16_t* __restrict__ Xl,
                                  const bf16_t* __restrict__ Vh,
                                  const bf16_t* __restrict__ Vl,
                                  const float* __restrict__ coeffs,
                                  float* __restrict__ out,
                                  int nrows) {
    const int wave = threadIdx.x >> 5;
    const int lane = threadIdx.x & 31;
    const int half = lane >> 4;      // 0: lanes 0-15, 1: lanes 16-31
    const int l15  = lane & 15;
    const int m0   = blockIdx.x * 16;
    const int o0   = wave * 16;

    v8f acc[NB];
#pragma unroll
    for (int b = 0; b < NB; ++b)
#pragma unroll
        for (int k = 0; k < 8; ++k) acc[b][k] = 0.0f;

    const bf16_t* xh = Xh + (size_t)(m0 + l15) * DD;
    const bf16_t* xl = Xl + (size_t)(m0 + l15) * DD;

#pragma unroll
    for (int kt = 0; kt < DD; kt += 32) {
        // ---- A tile: 16x32 bf16.  lane<16 -> K {kt..kt+15},
        //      lane>=16 -> K {kt+16..kt+31}; row = lane&15.
        const int ka = kt + half * 16;
        v8bf h0 = *(const v8bf*)(xh + ka);
        v8bf h1 = *(const v8bf*)(xh + ka + 8);
        v8bf l0 = *(const v8bf*)(xl + ka);
        v8bf l1 = *(const v8bf*)(xl + ka + 8);
        v16bf ah = __builtin_shufflevector(h0, h1, 0, 1, 2, 3, 4, 5, 6, 7,
                                           8, 9, 10, 11, 12, 13, 14, 15);
        v16bf al = __builtin_shufflevector(l0, l1, 0, 1, 2, 3, 4, 5, 6, 7,
                                           8, 9, 10, 11, 12, 13, 14, 15);

        // ---- B tiles: col = lane&15, K contiguous (pre-transposed V)
        const int kb = kt + half * 16;
#pragma unroll
        for (int b = 0; b < NB; ++b) {
            size_t voff = ((size_t)b * OO + o0 + l15) * DD + kb;
            v16bf bh = *(const v16bf*)(Vh + voff);
            v16bf bl = *(const v16bf*)(Vl + voff);
            // bf16x3: hi*hi + hi*lo + lo*hi, f32 accumulate
            acc[b] = __builtin_amdgcn_wmma_f32_16x16x32_bf16(
                false, ah, false, bh, (short)0, acc[b], false, false);
            acc[b] = __builtin_amdgcn_wmma_f32_16x16x32_bf16(
                false, ah, false, bl, (short)0, acc[b], false, false);
            acc[b] = __builtin_amdgcn_wmma_f32_16x16x32_bf16(
                false, al, false, bh, (short)0, acc[b], false, false);
        }
    }

    // ---- epilogue: combine bases per relation and store.
    // D layout: vgpr v -> M = m0 + v + 8*half ; col = o0 + (lane&15).
#pragma unroll
    for (int r = 0; r < NR; ++r) {
        v8f d;
        if (coeffs) {
#pragma unroll
            for (int k = 0; k < 8; ++k) d[k] = 0.0f;
#pragma unroll
            for (int b = 0; b < NB; ++b) {
                float c = coeffs[r * NB + b];
#pragma unroll
                for (int k = 0; k < 8; ++k) d[k] += c * acc[b][k];
            }
        } else {
            d = acc[0];
        }
        float* orow = out + ((size_t)r * nrows + m0 + half * 8) * OO + o0 + l15;
#pragma unroll
        for (int v = 0; v < 8; ++v) orow[(size_t)v * OO] = d[v];
    }
}

// ------------------------------------------------- edge gather / scatter
// One wave per edge: lane handles 4 of the 128 output floats.
__global__ void scatter_kernel(const float* __restrict__ xt,
                               const float* __restrict__ deg,
                               const int* __restrict__ rel,
                               const int* __restrict__ src,
                               const int* __restrict__ dst,
                               float* __restrict__ agg, int e) {
    int gid  = blockIdx.x * blockDim.x + threadIdx.x;
    int edge = gid >> 5;
    int lane = gid & 31;
    if (edge >= e) return;
    int re = rel[edge];
    int s  = src[edge];
    int dd = dst[edge];
    float w = 1.0f / fmaxf(deg[(size_t)re * NN + dd], 1.0f);
    const float* xp = xt + ((size_t)re * NN + s) * OO + lane * 4;
    float4 v = *(const float4*)xp;
    float* op = agg + (size_t)dd * OO + lane * 4;
    atomicAdd(op + 0, v.x * w);
    atomicAdd(op + 1, v.y * w);
    atomicAdd(op + 2, v.z * w);
    atomicAdd(op + 3, v.w * w);
}

// ------------------------------------------ fused self-loop + bias + relu
__global__ void final_kernel(const float* __restrict__ sl,
                             const float* __restrict__ bias,
                             float* __restrict__ out, int n) {
    int i = blockIdx.x * blockDim.x + threadIdx.x;
    if (i >= n) return;
    int o = i & (OO - 1);
    float v = out[i] + sl[i] + bias[o];
    out[i] = fmaxf(v, 0.0f);
}

// ---------------------------------------------------------------- launcher
extern "C" void kernel_launch(void* const* d_in, const int* in_sizes, int n_in,
                              void* d_out, int out_size, void* d_ws, size_t ws_size,
                              hipStream_t stream) {
    const float* x       = (const float*)d_in[0];
    const float* basis_v = (const float*)d_in[1];
    const float* coeffs  = (const float*)d_in[2];
    const float* w_loop  = (const float*)d_in[3];
    const float* bias_p  = (const float*)d_in[4];
    const int*   src     = (const int*)d_in[5];
    const int*   dst     = (const int*)d_in[6];
    const int*   rel     = (const int*)d_in[7];
    float* out = (float*)d_out;

    // workspace layout (256B aligned sections)
    char* ws = (char*)d_ws;
    float*  xt  = (float*)(ws + 0);                 // R*N*O f32 = 204,800,000 B
    float*  sl  = (float*)(ws + 204800000);         // N*O  f32 =  25,600,000 B
    float*  deg = (float*)(ws + 230400000);         // R*N  f32 =   1,600,000 B
    bf16_t* Xh  = (bf16_t*)(ws + 232000000);        // N*D bf16 =  12,800,000 B
    bf16_t* Xl  = (bf16_t*)(ws + 244800000);        // N*D bf16 =  12,800,000 B
    bf16_t* Bh  = (bf16_t*)(ws + 257600000);        // B*O*D bf16 =   131,072 B
    bf16_t* Bl  = (bf16_t*)(ws + 257731072);
    bf16_t* Lh  = (bf16_t*)(ws + 257862144);        // O*D bf16 =      32,768 B
    bf16_t* Ll  = (bf16_t*)(ws + 257894912);

    // 1) zero degree buffer and output accumulator
    fill_kernel<<<(RR * NN + 255) / 256, 256, 0, stream>>>(deg, 0.0f, RR * NN);
    fill_kernel<<<(NN * OO + 255) / 256, 256, 0, stream>>>(out, 0.0f, NN * OO);

    // 2) split X into bf16 hi/lo (one pass, removes cvt VALU from GEMM loop)
    xsplit_kernel<<<(NN * DD / 4 + 255) / 256, 256, 0, stream>>>(
        x, Xh, Xl, NN * DD / 4);

    // 3) split+transpose V bases and self-loop weight to bf16 hi/lo
    prep_kernel<<<dim3((DD * OO + 127) / 128, BB + 1), 128, 0, stream>>>(
        basis_v, w_loop, Bh, Bl, Lh, Ll);

    // 4) per-(relation,dst) in-degree
    deg_kernel<<<(EE + 255) / 256, 256, 0, stream>>>(rel, dst, deg, EE);

    // 5) basis-fused transform: xt[r] = sum_b coeffs[r,b] * (x @ V_b)
    gemm_basis_kernel<BB, RR><<<NN / 16, 256, 0, stream>>>(
        Xh, Xl, Bh, Bl, coeffs, xt, NN);

    // 6) self-loop GEMM: sl = x @ w_loop
    gemm_basis_kernel<1, 1><<<NN / 16, 256, 0, stream>>>(
        Xh, Xl, Lh, Ll, nullptr, sl, NN);

    // 7) normalized edge scatter (one wave per edge)
    scatter_kernel<<<(EE * 32 + 255) / 256, 256, 0, stream>>>(
        xt, deg, rel, src, dst, out, EE);

    // 8) fused self-loop + bias + relu
    final_kernel<<<(NN * OO + 255) / 256, 256, 0, stream>>>(sl, bias_p, out, NN * OO);
}
